// Net_37048387896011
// MI455X (gfx1250) — matrix operand order repaired and checked by reading, not compiled, not run on hardware
//
#include <hip/hip_runtime.h>
#include <hip/hip_bf16.h>

// ---------------- geometry ----------------
#define IMG     1536
#define TILE_H  16
#define TILE_W  64
#define WP      80          // padded stage width (5 groups of 16)
#define H0      22          // s0 rows (halo 3)
#define H1      20          // s1 rows (halo 2)
#define H2      18          // s2 rows (halo 1)
#define CH      16          // channels per stage (conv1 Cin zero-padded to 16)

#define OFF_A   0
#define SZ_A    (H0 * WP * CH)        // 28160 f16 (s0, reused as s2)
#define OFF_B   (OFF_A + SZ_A)
#define SZ_B    (H1 * WP * CH)        // 25600 f16 (s1)
#define OFF_W   (OFF_B + SZ_B)        // weights: [layer][tap(10)][co(16)][ci(16)]
#define SZ_W    (3 * 10 * 16 * 16)    // 7680 f16
#define LDS_TOT (OFF_W + SZ_W + 64)   // + tail pad; ~123 KB total

#define NTHREADS 256
#define NWAVES   8

typedef _Float16 h8   __attribute__((ext_vector_type(8)));
typedef _Float16 v16h __attribute__((ext_vector_type(16)));
typedef float    v8f  __attribute__((ext_vector_type(8)));

__global__ __launch_bounds__(NTHREADS)
void conv3x3x3_fused_wmma(const float* __restrict__ x,
                          const float* __restrict__ w1,
                          const float* __restrict__ w2,
                          const float* __restrict__ w3,
                          float* __restrict__ out)
{
    __shared__ _Float16 smem[LDS_TOT];

    const int tid  = threadIdx.x;
    const int lane = tid & 31;
    // Wave index is wave-uniform; pin it to an SGPR so the job loop becomes a
    // scalar loop (s_cmp/s_cbranch) instead of an exec-masked vector loop.
    const int wave = __builtin_amdgcn_readfirstlane(tid >> 5);
    const int tx   = blockIdx.x;            // 0..23
    const int ty   = blockIdx.y;            // 0..95
    const int b    = blockIdx.z;            // 0..3
    const int y0   = ty * TILE_H;
    const int xg0  = tx * TILE_W;

    // ---- stage weights: w[l][tap][co][ci], tap 9 = zeros, conv1 ci>=3 = zeros ----
    for (int i = tid; i < SZ_W; i += NTHREADS) {
        int ci  = i & 15;
        int co  = (i >> 4) & 15;
        int tap = (i >> 8) % 10;
        int l   = i / 2560;
        float w = 0.f;
        if (tap < 9) {
            int dy = tap / 3, dx = tap % 3;
            if (l == 0) {
                if (ci < 3) w = w1[((co * 3 + ci) * 3 + dy) * 3 + dx];
            } else {
                const float* wp = (l == 1) ? w2 : w3;
                w = wp[((co * 16 + ci) * 3 + dy) * 3 + dx];
            }
        }
        smem[OFF_W + i] = (_Float16)w;
    }

    // ---- stage input tile into s0: channel-last f16, zero padded ----
    for (int p = tid; p < H0 * WP; p += NTHREADS) {
        int r = p / WP, c = p % WP;
        int Y = y0 - 3 + r;
        int X = xg0 - 3 + c;
        float f0 = 0.f, f1 = 0.f, f2 = 0.f;
        if ((unsigned)Y < (unsigned)IMG && (unsigned)X < (unsigned)IMG && c < (TILE_W + 6)) {
            size_t o = ((size_t)(b * 3) * IMG + Y) * IMG + X;
            f0 = x[o];
            f1 = x[o + (size_t)IMG * IMG];
            f2 = x[o + 2 * (size_t)IMG * IMG];
        }
        h8 lo, hi;
        #pragma unroll
        for (int i = 0; i < 8; ++i) { lo[i] = (_Float16)0.f; hi[i] = (_Float16)0.f; }
        lo[0] = (_Float16)f0; lo[1] = (_Float16)f1; lo[2] = (_Float16)f2;
        *(h8*)&smem[OFF_A + p * CH]     = lo;   // ci 0..7
        *(h8*)&smem[OFF_A + p * CH + 8] = hi;   // ci 8..15 (zero)
    }
    __syncthreads();

    const int n   = lane & 15;               // D/B column (pixel in group)
    const int klo = (lane < 16) ? 0 : 8;     // A: ci half per lane-half
    const int bth = lane >> 4;               // B: tap-in-chunk per lane-half
    const int mb  = (lane < 16) ? 0 : 8;     // D: M base per lane-half

    // Per-lane tap offsets (loop invariant across jobs & layers):
    // chunk ch uses tap t = 2*ch + bth; tap 9 aliases tap 8 (its weights are 0).
    int bofs[5];
    #pragma unroll
    for (int ch = 0; ch < 5; ++ch) {
        int t  = 2 * ch + bth;
        int dy = (t == 9) ? 2 : (t / 3);
        int dx = (t == 9) ? 2 : (t % 3);
        bofs[ch] = (dy * WP + dx) * CH;
    }

    #pragma unroll
    for (int layer = 0; layer < 3; ++layer) {
        const int srcOff = (layer == 1) ? OFF_B : OFF_A;   // s0, s1, s2(=s0)
        const int dstOff = (layer == 0) ? OFF_B : OFF_A;
        const int rows   = H1 - 2 * layer;                 // 20,18,16
        const int groups = (layer == 2) ? 4 : 5;

        // Preload A (weights) into registers: A[m][k], k = tapPair*16 + ci.
        // Lane layout (contiguous k order chosen here, as for B): lanes 0-15: K0-15 (tap t0, all ci);
        // lanes 16-31: K16-31 (tap t1, all ci) -> ci 0..7 and 8..15 of tap t0+bth.
        v16h aw[5];
        #pragma unroll
        for (int ch = 0; ch < 5; ++ch) {
            int t0 = 2 * ch, t1 = t0 + 1;
            h8 alo = *(const h8*)&smem[OFF_W + ((layer * 10 + t0 + bth) * 16 + n) * 16];
            h8 ahi = *(const h8*)&smem[OFF_W + ((layer * 10 + t0 + bth) * 16 + n) * 16 + 8];
            aw[ch] = __builtin_shufflevector(alo, ahi,
                     0, 1, 2, 3, 4, 5, 6, 7, 8, 9, 10, 11, 12, 13, 14, 15);
        }

        const int nJobs = rows * groups;
        for (int j = wave; j < nJobs; j += NWAVES) {
            const int row   = j / groups;
            const int gx    = (j - row * groups) * 16;
            const int pbase = srcOff + (row * WP + gx + n) * CH;

            // Issue all B loads for this job first (10x ds_load_b128), then
            // run the 5 chained WMMAs so loads overlap the accumulation chain.
            // B layout (contiguous k order chosen here, as for A): lanes 0-15 hold K0-15 (tap t0, all ci),
            // lanes 16-31 hold K16-31 (tap t1, all ci) of column n.
            v16h bv[5];
            #pragma unroll
            for (int ch = 0; ch < 5; ++ch) {
                const int base = pbase + bofs[ch];
                h8 blo = *(const h8*)&smem[base];
                h8 bhi = *(const h8*)&smem[base + 8];
                bv[ch] = __builtin_shufflevector(blo, bhi,
                         0, 1, 2, 3, 4, 5, 6, 7, 8, 9, 10, 11, 12, 13, 14, 15);
            }

            v8f acc;
            #pragma unroll
            for (int i = 0; i < 8; ++i) acc[i] = 0.f;
            #pragma unroll
            for (int ch = 0; ch < 5; ++ch) {
                acc = __builtin_amdgcn_wmma_f32_16x16x32_f16(
                          false, aw[ch], false, bv[ch], (short)0, acc, false, false);
            }

            if (layer < 2) {
                // D 16x16 f32: lane holds M = mb + r, N = n. Write channel-last f16.
                h8 o;
                #pragma unroll
                for (int r8 = 0; r8 < 8; ++r8) o[r8] = (_Float16)acc[r8];
                *(h8*)&smem[dstOff + (row * WP + gx + n) * CH + mb] = o;
            } else {
                // Final conv: ReLU + store f32 NCHW.
                int Y = y0 + row;
                int X = xg0 + gx + n;
                size_t gbase = ((size_t)(b * 16 + mb) * IMG + Y) * IMG + X;
                #pragma unroll
                for (int r8 = 0; r8 < 8; ++r8)
                    out[gbase + (size_t)r8 * IMG * IMG] = fmaxf(acc[r8], 0.f);
            }
        }
        __syncthreads();
    }
}

extern "C" void kernel_launch(void* const* d_in, const int* in_sizes, int n_in,
                              void* d_out, int out_size, void* d_ws, size_t ws_size,
                              hipStream_t stream) {
    (void)in_sizes; (void)n_in; (void)out_size; (void)d_ws; (void)ws_size;
    const float* x  = (const float*)d_in[0];
    const float* w1 = (const float*)d_in[1];
    const float* w2 = (const float*)d_in[2];
    const float* w3 = (const float*)d_in[3];
    float* out = (float*)d_out;

    dim3 grid(IMG / TILE_W, IMG / TILE_H, 4);   // 24 x 96 x 4 tiles
    conv3x3x3_fused_wmma<<<grid, NTHREADS, 0, stream>>>(x, w1, w2, w3, out);
}
